// MultiHeadAttention_32315333935906
// MI455X (gfx1250) — matrix operand order repaired and checked by reading, not compiled, not run on hardware
//
#include <hip/hip_runtime.h>
#include <hip/hip_bf16.h>

// ---------------------------------------------------------------------------
// MI455X (gfx1250) multi-head causal attention, bf16 WMMA pipeline.
//   B=2, T=2048, C=1024, H=16, HD=64
// cvt(x)->bf16 ; W -> bf16 N-major ; GEMM (async-LDS staged B, double
// buffered, 8 wmma/iter per wave) ; V transpose ; flash attention (S^T trick,
// pipelined loads) ; GEMM2 (+bias). Matrix math: V_WMMA_F32_16X16X32_BF16.
// ---------------------------------------------------------------------------

typedef __attribute__((ext_vector_type(16))) __bf16 v16bf;
typedef __attribute__((ext_vector_type(8)))  __bf16 v8bf;
typedef __attribute__((ext_vector_type(4)))  __bf16 v4bf;
typedef __attribute__((ext_vector_type(8)))  float  v8f;
typedef __attribute__((ext_vector_type(4)))  int    v4i;

// address-space qualified v4i for the async builtin's pointer params
typedef __attribute__((address_space(1))) v4i g_v4i;  // global
typedef __attribute__((address_space(3))) v4i l_v4i;  // LDS

#define DEVINL __device__ __forceinline__
#define FINF __builtin_inff()

#define T_SEQ 2048
#define NH    16
#define HDIM  64
#define CDIM  1024
#define C3    3072
#define BB    2

// ---- async global -> LDS (gfx1250) ----------------------------------------
#if __has_builtin(__builtin_amdgcn_global_load_async_to_lds_b128)
#define HAVE_ASYNC_LDS 1
#else
#define HAVE_ASYNC_LDS 0
#pragma message("gfx1250 async global->LDS builtin: MISSING, sync fallback")
#endif

DEVINL void async_copy16(const __bf16* __restrict__ g, __bf16* l) {
#if HAVE_ASYNC_LDS
  __builtin_amdgcn_global_load_async_to_lds_b128((g_v4i*)g, (l_v4i*)l, 0, 0);
#else
  *(v8bf*)l = *(const v8bf*)g;
#endif
}

DEVINL void wait_async_le1() {
#if HAVE_ASYNC_LDS
#if __has_builtin(__builtin_amdgcn_s_wait_asynccnt)
  __builtin_amdgcn_s_wait_asynccnt(1);
#else
  asm volatile("s_wait_asynccnt 1" ::: "memory");
#endif
#endif
}

// ---- WMMA fragment loaders -------------------------------------------------
// A (16x32 bf16, row-major source): lane L: row m0+(L&15), h=L>>4;
//   elements 0..7 = K[k0+8h ..], elements 8..15 = K[k0+16+8h ..]
DEVINL v16bf load_a_frag(const __bf16* __restrict__ A, int lda, int m0, int k0,
                         int lane) {
  const int row = m0 + (lane & 15);
  const int h   = lane >> 4;
  const __bf16* p = A + (size_t)row * lda + k0 + 8 * h;
  v8bf lo = *(const v8bf*)(p);
  v8bf hi = *(const v8bf*)(p + 16);
  v16bf r;
#pragma unroll
  for (int i = 0; i < 8; ++i) { r[i] = lo[i]; r[i + 8] = hi[i]; }
  return r;
}

// B (32x16 bf16) from N-major storage Bt[n][k]: lane L: col n0+(L&15),
// elements 0..7 = K[k0+8h ..], elements 8..15 = K[k0+16+8h ..]
DEVINL v16bf load_b_frag(const __bf16* __restrict__ Bt, int ldb, int n0, int k0,
                         int lane) {
  const int col = n0 + (lane & 15);
  const int h   = lane >> 4;
  const __bf16* p = Bt + (size_t)col * ldb + k0 + 8 * h;
  v8bf lo = *(const v8bf*)(p);
  v8bf hi = *(const v8bf*)(p + 16);
  v16bf r;
#pragma unroll
  for (int i = 0; i < 8; ++i) { r[i] = lo[i]; r[i + 8] = hi[i]; }
  return r;
}

DEVINL v8f wmma_bf16(v16bf a, v16bf b, v8f c) {
  return __builtin_amdgcn_wmma_f32_16x16x32_bf16(false, a, false, b, (short)0,
                                                 c, false, false);
}

// ---- f32 -> bf16 elementwise convert (vec4) --------------------------------
__global__ __launch_bounds__(256) void cvt_kernel(const float* __restrict__ x,
                                                  __bf16* __restrict__ xb,
                                                  int n4) {
  int i = blockIdx.x * 256 + threadIdx.x;
  if (i < n4) {
    const float4 f = ((const float4*)x)[i];
    v4bf o;
    o[0] = (__bf16)f.x; o[1] = (__bf16)f.y;
    o[2] = (__bf16)f.z; o[3] = (__bf16)f.w;
    ((v4bf*)xb)[i] = o;
  }
}

// ---- W [K][N] f32 -> Wt [N][K] bf16 (LDS tile transpose) -------------------
__global__ __launch_bounds__(256) void wtrans_kernel(
    const float* __restrict__ W, __bf16* __restrict__ Wt, int K, int N) {
  __shared__ float tile[32][33];
  const int n0 = blockIdx.x * 32, k0 = blockIdx.y * 32;
  const int tx = threadIdx.x, ty = threadIdx.y;
#pragma unroll
  for (int i = 0; i < 4; ++i)
    tile[ty + i * 8][tx] = W[(size_t)(k0 + ty + i * 8) * N + n0 + tx];
  __syncthreads();
#pragma unroll
  for (int i = 0; i < 4; ++i)
    Wt[(size_t)(n0 + ty + i * 8) * K + k0 + tx] = (__bf16)tile[tx][ty + i * 8];
}

// ---- V slice of qkv -> Vt[b][h][d][t] (bf16, LDS transpose) ----------------
__global__ __launch_bounds__(256) void vtrans_kernel(
    const __bf16* __restrict__ qkv, __bf16* __restrict__ vt) {
  __shared__ __bf16 tile[64][72];
  const int bh = blockIdx.x;
  const int b = bh >> 4, head = bh & 15;
  const int t0 = blockIdx.y * 64;
  const int tid = threadIdx.x;
#pragma unroll
  for (int i = 0; i < 16; ++i) {
    int idx = tid + i * 256;
    int tl = idx >> 6, d = idx & 63;
    tile[tl][d] =
        qkv[((size_t)b * T_SEQ + t0 + tl) * C3 + 2 * CDIM + head * HDIM + d];
  }
  __syncthreads();
#pragma unroll
  for (int i = 0; i < 16; ++i) {
    int idx = tid + i * 256;
    int d = idx >> 6, tl = idx & 63;
    vt[((size_t)bh * HDIM + d) * T_SEQ + t0 + tl] = tile[tl][d];
  }
}

// ---- bf16 WMMA GEMM: C[M][N] = A[M][K] * Bt[N][K]^T (+bias) ----------------
// Block 256 thr = 8 waves; block tile 256(M) x 64(N); wave tile 32 x 64
// (2 M-tiles x 4 N-tiles = 8 wmma per k-step). Shared B panel staged in LDS
// via async global->LDS copies, double buffered. Row stride 40 bf16 (80 B =
// 20 dwords) keeps the ds_load_b128 fragment reads bank-conflict-free.
#define BROW 40
template <bool OUT_BF16>
__global__ __launch_bounds__(256) void gemm_bf16_kernel(
    const __bf16* __restrict__ A, const __bf16* __restrict__ Bt,
    __bf16* __restrict__ Cb, float* __restrict__ Cf,
    const float* __restrict__ bias, int M, int N, int K) {
  __shared__ __bf16 bpanel[2][64 * BROW];
  const int tid = threadIdx.x;
  const int lane = tid & 31;
  const int wave = tid >> 5;
  const int m0 = blockIdx.y * 256 + wave * 32;
  const int n0 = blockIdx.x * 64;

  // stage one 64(n) x 32(k) B panel: 256 threads x 16 bytes
  const int srow = tid >> 2;          // 0..63
  const int sseg = (tid & 3) * 8;     // bf16 offset of 16B segment
  auto stage = [&](int buf, int k0) {
    async_copy16(Bt + (size_t)(n0 + srow) * K + k0 + sseg,
                 &bpanel[buf][srow * BROW + sseg]);
  };

  const int nk = K / 32;
  stage(0, 0);
  stage(1, 32);
  v16bf a_cur0 = load_a_frag(A, K, m0, 0, lane);
  v16bf a_cur1 = load_a_frag(A, K, m0 + 16, 0, lane);
  v8f acc[2][4] = {};

  for (int i = 0; i < nk; ++i) {
    wait_async_le1();          // panel i complete (in-order), i+1 may fly
    __syncthreads();

    const __bf16* bp = bpanel[i & 1];
    const int h = lane >> 4;
    v16bf b[4];
#pragma unroll
    for (int nt = 0; nt < 4; ++nt) {
      const __bf16* q = bp + (nt * 16 + (lane & 15)) * BROW + 8 * h;
      v8bf lo = *(const v8bf*)(q);
      v8bf hi = *(const v8bf*)(q + 16);
#pragma unroll
      for (int e = 0; e < 8; ++e) { b[nt][e] = lo[e]; b[nt][e + 8] = hi[e]; }
    }

    v16bf a_next0 = a_cur0, a_next1 = a_cur1;
    if (i + 1 < nk) {
      a_next0 = load_a_frag(A, K, m0, (i + 1) * 32, lane);
      a_next1 = load_a_frag(A, K, m0 + 16, (i + 1) * 32, lane);
      __builtin_prefetch(A + (size_t)(m0 + (lane & 15)) * K + (i + 2) * 32, 0,
                         1);
    }

#pragma unroll
    for (int nt = 0; nt < 4; ++nt)
      acc[0][nt] = wmma_bf16(a_cur0, b[nt], acc[0][nt]);
#pragma unroll
    for (int nt = 0; nt < 4; ++nt)
      acc[1][nt] = wmma_bf16(a_cur1, b[nt], acc[1][nt]);
    a_cur0 = a_next0;
    a_cur1 = a_next1;

    __syncthreads();           // all waves done reading panel i
    if (i + 2 < nk) stage(i & 1, (i + 2) * 32);
  }

  const int h = lane >> 4;
  const int nq = lane & 15;
#pragma unroll
  for (int mt = 0; mt < 2; ++mt) {
#pragma unroll
    for (int nt = 0; nt < 4; ++nt) {
#pragma unroll
      for (int j = 0; j < 8; ++j) {
        int row = m0 + mt * 16 + j + 8 * h;
        int col = n0 + nt * 16 + nq;
        float v = acc[mt][nt][j];
        if (bias) v += bias[col];
        if (OUT_BF16)
          Cb[(size_t)row * N + col] = (__bf16)v;
        else
          Cf[(size_t)row * N + col] = v;
      }
    }
  }
}

// ---- Flash attention -------------------------------------------------------
// qkv row layout (3072): [K | Q | V] (reference splits k first). Each wave:
// 16 queries x HD=64; S^T = K*Q^T so softmax is per-lane + one shfl_xor(16);
// S^T f32 fragments convert in-place into the P bf16 A-fragment.
__global__ __launch_bounds__(256) void attn_kernel(
    const __bf16* __restrict__ qkv, const __bf16* __restrict__ vt,
    __bf16* __restrict__ attnout) {
  const int lane = threadIdx.x & 31;
  const int wave = threadIdx.x >> 5;
  const int bh = blockIdx.x;  // b*16 + head
  const int b = bh >> 4, head = bh & 15;
  const int q0 = blockIdx.y * 128 + wave * 16;
  const int h = lane >> 4;
  const int nq = lane & 15;

  const __bf16* kbase = qkv + (size_t)b * T_SEQ * C3 + head * HDIM;  // K rows
  const __bf16* qbase = kbase + CDIM;                                // Q rows
  const __bf16* vtb = vt + (size_t)bh * HDIM * T_SEQ;  // Vt rows (d-major)

  // Q as N-major B operand: resident across the whole key loop.
  v16bf qb0 = load_b_frag(qbase, C3, q0, 0, lane);
  v16bf qb1 = load_b_frag(qbase, C3, q0, 32, lane);

  v8f acc[4] = {};
  float m_run = -FINF, l_run = 0.f;

  const int kend = q0 + 16;  // causal: keys < q0+16
  for (int kt = 0; kt < kend; kt += 32) {
    // Issue ALL loads for this chunk up front (forms clauses, overlaps wmma).
    v16bf ka0 = load_a_frag(kbase, C3, kt, 0, lane);
    v16bf ka1 = load_a_frag(kbase, C3, kt, 32, lane);
    v16bf ka2 = load_a_frag(kbase, C3, kt + 16, 0, lane);
    v16bf ka3 = load_a_frag(kbase, C3, kt + 16, 32, lane);
    v16bf vb[4];
#pragma unroll
    for (int nt = 0; nt < 4; ++nt)
      vb[nt] = load_b_frag(vtb, T_SEQ, nt * 16, kt, lane);
    if (kt + 32 < kend)
      __builtin_prefetch(kbase + (size_t)(kt + 32 + (lane & 15)) * C3, 0, 1);

    v8f s0 = {}, s1 = {};
    s0 = wmma_bf16(ka0, qb0, s0);
    s0 = wmma_bf16(ka1, qb1, s0);
    s1 = wmma_bf16(ka2, qb0, s1);
    s1 = wmma_bf16(ka3, qb1, s1);

    const int q = q0 + nq;
    float x0[8], x1[8];
    float cm = -FINF;
#pragma unroll
    for (int j = 0; j < 8; ++j) {
      int key0 = kt + j + 8 * h;
      int key1 = key0 + 16;
      x0[j] = (key0 <= q) ? s0[j] * 0.125f : -FINF;  // HD^-0.5 = 0.125
      x1[j] = (key1 <= q) ? s1[j] * 0.125f : -FINF;
      cm = fmaxf(cm, fmaxf(x0[j], x1[j]));
    }
    cm = fmaxf(cm, __shfl_xor(cm, 16, 32));  // merge other 16 keys
    const float m_new = fmaxf(m_run, cm);
    const float alpha = __expf(m_run - m_new);

    float ssum = 0.f;
    v16bf pa;  // P bf16 A-fragment: elems 0..7 <- tile0, 8..15 <- tile1
#pragma unroll
    for (int j = 0; j < 8; ++j) {
      float e0 = __expf(x0[j] - m_new);
      float e1 = __expf(x1[j] - m_new);
      ssum += e0 + e1;
      pa[j] = (__bf16)e0;
      pa[j + 8] = (__bf16)e1;
    }
    ssum += __shfl_xor(ssum, 16, 32);
    l_run = l_run * alpha + ssum;
    m_run = m_new;

    // Rescale O accumulators: row of O fragment = query j+8h -> broadcast.
#pragma unroll
    for (int j = 0; j < 8; ++j) {
      float aj = __shfl(alpha, j + 8 * h, 32);
#pragma unroll
      for (int nt = 0; nt < 4; ++nt) acc[nt][j] *= aj;
    }

#pragma unroll
    for (int nt = 0; nt < 4; ++nt) acc[nt] = wmma_bf16(pa, vb[nt], acc[nt]);
  }

  const float linv = 1.0f / l_run;
#pragma unroll
  for (int j = 0; j < 8; ++j) {
    float lj = __shfl(linv, j + 8 * h, 32);
    int row = q0 + j + 8 * h;
#pragma unroll
    for (int nt = 0; nt < 4; ++nt) {
      attnout[((size_t)b * T_SEQ + row) * CDIM + head * HDIM + nt * 16 + nq] =
          (__bf16)(acc[nt][j] * lj);
    }
  }
}

// ---------------------------------------------------------------------------
extern "C" void kernel_launch(void* const* d_in, const int* in_sizes, int n_in,
                              void* d_out, int out_size, void* d_ws,
                              size_t ws_size, hipStream_t stream) {
  const float* x  = (const float*)d_in[0];
  const float* Wa = (const float*)d_in[1];
  const float* Wp = (const float*)d_in[2];
  const float* bp = (const float*)d_in[3];
  float* out = (float*)d_out;

  char* ws = (char*)d_ws;
  size_t off = 0;
  auto alloc = [&](size_t bytes) {
    void* p = ws + off;
    off = (off + bytes + 255) & ~(size_t)255;
    return p;
  };
  __bf16* xb  = (__bf16*)alloc((size_t)BB * T_SEQ * CDIM * 2);       //  8 MiB
  __bf16* Wta = (__bf16*)alloc((size_t)C3 * CDIM * 2);               //  6 MiB
  __bf16* Wtp = (__bf16*)alloc((size_t)CDIM * CDIM * 2);             //  2 MiB
  __bf16* qkv = (__bf16*)alloc((size_t)BB * T_SEQ * C3 * 2);         // 24 MiB
  __bf16* vtb = (__bf16*)alloc((size_t)BB * NH * HDIM * T_SEQ * 2);  //  8 MiB
  __bf16* att = (__bf16*)alloc((size_t)BB * T_SEQ * CDIM * 2);       //  8 MiB

  cvt_kernel<<<(BB * T_SEQ * CDIM / 4 + 255) / 256, 256, 0, stream>>>(
      x, xb, BB * T_SEQ * CDIM / 4);
  wtrans_kernel<<<dim3(C3 / 32, CDIM / 32), dim3(32, 8), 0, stream>>>(
      Wa, Wta, CDIM, C3);
  wtrans_kernel<<<dim3(CDIM / 32, CDIM / 32), dim3(32, 8), 0, stream>>>(
      Wp, Wtp, CDIM, CDIM);
  gemm_bf16_kernel<true><<<dim3(C3 / 64, BB * T_SEQ / 256), 256, 0, stream>>>(
      xb, Wta, qkv, nullptr, nullptr, BB * T_SEQ, C3, CDIM);
  vtrans_kernel<<<dim3(BB * NH, T_SEQ / 64), 256, 0, stream>>>(qkv, vtb);
  attn_kernel<<<dim3(BB * NH, T_SEQ / 128), 256, 0, stream>>>(qkv, vtb, att);
  gemm_bf16_kernel<false><<<dim3(CDIM / 64, BB * T_SEQ / 256), 256, 0,
                            stream>>>(att, Wtp, nullptr, out, bp, BB * T_SEQ,
                                      CDIM, CDIM);
}
